// ShallowSplineConvNet_16561393893731
// MI455X (gfx1250) — matrix operand order repaired and checked by reading, not compiled, not run on hardware
//
#include <hip/hip_runtime.h>

#define N_NODES   32492
#define N_EDGES   196608
#define KS        25
#define KTOT      15625
#define S_PROD    8
#define PAIRS     (N_EDGES * S_PROD)          // 1,572,864
#define MAX_TILES (PAIRS / 16 + KTOT)         // 113,929 upper bound
#define MTILES    ((N_NODES + 15) / 16)       // 2031
#define BN_EPS    1e-5f

typedef __attribute__((ext_vector_type(16))) __bf16   v16bf;
typedef __attribute__((ext_vector_type(8)))  __bf16   v8bf;
typedef __attribute__((ext_vector_type(8)))  float    v8f;
typedef __attribute__((ext_vector_type(8)))  unsigned v8u;

// ---------- compile-time loop ----------
template<int V> struct ic { static constexpr int value = V; };
template<int N, int I = 0, typename F>
__device__ __forceinline__ void static_for(F&& f) {
  if constexpr (I < N) { f(ic<I>{}); static_for<N, I + 1>(f); }
}

// ---------- bf16 helper (round-to-nearest-even), scalar uses only ----------
__device__ __forceinline__ __bf16 f2bf(float f) {
  unsigned u = __builtin_bit_cast(unsigned, f);
  unsigned r = u + 0x7FFFu + ((u >> 16) & 1u);
  unsigned short h = (unsigned short)(r >> 16);
  return __builtin_bit_cast(__bf16, h);
}

// ---------- 1) open B-spline basis (degree 1) + flattened weight index ----------
__global__ void basis_kernel(const float* __restrict__ attr,
                             int* __restrict__ wi, float* __restrict__ bas) {
  int e = blockIdx.x * blockDim.x + threadIdx.x;
  if (e >= N_EDGES) return;
  float fr[3]; int iv[3];
#pragma unroll
  for (int d = 0; d < 3; ++d) {
    float v  = attr[e * 3 + d] * (float)(KS - 1);
    float vf = floorf(v);
    fr[d] = v - vf;
    iv[d] = (int)vf;
  }
#pragma unroll
  for (int s = 0; s < S_PROD; ++s) {
    float b = 1.0f; int w = 0; int mult = 1;
#pragma unroll
    for (int d = 0; d < 3; ++d) {
      int bit = (s >> d) & 1;
      b *= bit ? fr[d] : (1.0f - fr[d]);
      int id = (iv[d] + bit) % KS;
      w += id * mult;
      mult *= KS;
    }
    wi[e * S_PROD + s]  = w;
    bas[e * S_PROD + s] = b;
  }
}

// ---------- 2) counting sort of (edge,s) pairs by kernel index ----------
__global__ void hist_kernel(const int* __restrict__ wi, int* __restrict__ cnt) {
  int p = blockIdx.x * blockDim.x + threadIdx.x;
  if (p >= PAIRS) return;
  atomicAdd(&cnt[wi[p]], 1);
}

__global__ void scan_kernel(const int* __restrict__ cnt,
                            int* __restrict__ bin_start,
                            int* __restrict__ tile_start,
                            int* __restrict__ n_tiles_out) {
  __shared__ int sp[1024];
  __shared__ int st[1024];
  int t = threadIdx.x;
  int s = 0, s2 = 0;
  for (int i = 0; i < 16; ++i) {
    int idx = t * 16 + i;
    if (idx < KTOT) { int c = cnt[idx]; s += c; s2 += (c + 15) >> 4; }
  }
  sp[t] = s; st[t] = s2;
  __syncthreads();
  if (t == 0) {
    int a = 0, b = 0;
    for (int i = 0; i < 1024; ++i) {
      int x = sp[i]; sp[i] = a; a += x;
      int y = st[i]; st[i] = b; b += y;
    }
    n_tiles_out[0] = b;
  }
  __syncthreads();
  int off = sp[t], off2 = st[t];
  for (int i = 0; i < 16; ++i) {
    int idx = t * 16 + i;
    if (idx < KTOT) {
      int c = cnt[idx];
      bin_start[idx]  = off;  off  += c;
      tile_start[idx] = off2; off2 += (c + 15) >> 4;
    }
  }
}

__global__ void tile_fill_kernel(const int* __restrict__ cnt,
                                 const int* __restrict__ bin_start,
                                 const int* __restrict__ tile_start,
                                 int* __restrict__ tileK, int* __restrict__ tileP) {
  int b = blockIdx.x * blockDim.x + threadIdx.x;
  if (b >= KTOT) return;
  int c  = cnt[b];
  int nt = (c + 15) >> 4;
  int ts = tile_start[b];
  int ps = bin_start[b];
  for (int t = 0; t < nt; ++t) {
    int rem = c - 16 * t; if (rem > 16) rem = 16;
    tileK[ts + t] = b;
    tileP[ts + t] = (ps + 16 * t) | (rem << 24);   // start fits in 21 bits
  }
}

__global__ void scatter_sort_kernel(const int* __restrict__ wi,
                                    const float* __restrict__ bas,
                                    const int* __restrict__ bin_start,
                                    int* __restrict__ bin_cur,
                                    int* __restrict__ pair_edge,
                                    float* __restrict__ pair_bas) {
  int p = blockIdx.x * blockDim.x + threadIdx.x;
  if (p >= PAIRS) return;
  int k = wi[p];
  int pos = bin_start[k] + atomicAdd(&bin_cur[k], 1);
  pair_edge[pos] = p >> 3;      // edge id
  pair_bas[pos]  = bas[p];
}

// ---------- 3) convert input features to padded bf16 ----------
__global__ void convert_x_kernel(const float* __restrict__ x, int fin, int kstride,
                                 __bf16* __restrict__ out) {
  int idx = blockIdx.x * blockDim.x + threadIdx.x;
  if (idx >= N_NODES * kstride) return;
  int n = idx / kstride, k = idx - n * kstride;
  out[idx] = f2bf(k < fin ? x[n * fin + k] : 0.0f);
}

// ---------- A-fragment: two contiguous 16B loads per lane (ISA 7.12.2 layout) ----------
__device__ __forceinline__ v16bf load_a_frag(const __bf16* __restrict__ rowp,
                                             int kk, int abase) {
  const __bf16* ap = rowp + kk + abase;
  v8bf lo = *(const v8bf*)(ap);        // K = kk+abase   .. +7
  v8bf hi = *(const v8bf*)(ap + 16);   // K = kk+abase+16.. +23
  v16bf a;
#pragma unroll
  for (int j = 0; j < 8; ++j) { a[j] = lo[j]; a[j + 8] = hi[j]; }
  return a;
}

// ---------- B-fragment: per-lane column read; interior tiles have NO clamps or
//            selects (constexpr-stripped); f32->bf16 via one v_perm_b32/dword ----
template<int FIN, int FOUT, int KK, int NBASE>
__device__ __forceinline__ v16bf load_b_frag(const float* __restrict__ Wk,
                                             int koff, int col) {
  constexpr bool kfull = (KK + 31 < FIN);    // covers koff in {0,8}, j in [0,15]
  constexpr bool nfull = (NBASE + 15 < FOUT);
  const int cc  = NBASE + col;
  const int ccc = nfull ? cc : (cc < FOUT ? cc : FOUT - 1);
  v8u bu;
#pragma unroll
  for (int jj = 0; jj < 8; ++jj) {
    const int j0  = 2 * jj, j1 = j0 + 1;
    const int ki0 = KK + koff + j0 + (j0 >= 8 ? 8 : 0);
    const int ki1 = KK + koff + j1 + (j1 >= 8 ? 8 : 0);
    const int kc0 = kfull ? ki0 : (ki0 < FIN ? ki0 : FIN - 1);
    const int kc1 = kfull ? ki1 : (ki1 < FIN ? ki1 : FIN - 1);
    unsigned u0 = __builtin_bit_cast(unsigned, Wk[kc0 * FOUT + ccc]); // in-bounds
    unsigned u1 = __builtin_bit_cast(unsigned, Wk[kc1 * FOUT + ccc]);
    if constexpr (!(kfull && nfull)) {
      bool ok0 = (kfull || ki0 < FIN) && (nfull || cc < FOUT);
      bool ok1 = (kfull || ki1 < FIN) && (nfull || cc < FOUT);
      u0 = ok0 ? u0 : 0u;
      u1 = ok1 ? u1 : 0u;
    }
    u0 += 0x8000u;                                    // round-half-up to bf16
    u1 += 0x8000u;
    // {hi16(u1), hi16(u0)}: low half = elem j0, high = elem j1
    bu[jj] = __builtin_amdgcn_perm(u1, u0, 0x07060302u);
  }
  return __builtin_bit_cast(v16bf, bu);
}

// ---------- 4a) root GEMM: out = hb @ root + bias (dense, WMMA) ----------
template<int KTILES, int NTILES, int FIN, int FOUT, int KSTRIDE>
__global__ void root_gemm_t(const __bf16* __restrict__ hb,
                            const float* __restrict__ rootW,
                            const float* __restrict__ bias,
                            float* __restrict__ out) {
  const int lane = threadIdx.x;
  const int nb   = blockIdx.x * 16;
  const int row  = lane & 15;
  int node = nb + row; if (node >= N_NODES) node = N_NODES - 1;  // clamp, mask at store
  const __bf16* rowp = hb + (long)node * KSTRIDE;
  const int abase = (lane < 16) ? 0 : 8;
  const int koff  = (lane < 16) ? 0 : 8;
  const int col   = lane & 15;

  v8f acc[NTILES];
#pragma unroll
  for (int nt = 0; nt < NTILES; ++nt) acc[nt] = {};

  static_for<KTILES>([&](auto KT_) {
    constexpr int KK = decltype(KT_)::value * 32;
    v16bf a = load_a_frag(rowp, KK, abase);
    static_for<NTILES>([&](auto NT_) {
      constexpr int NT = decltype(NT_)::value;
      v16bf b = load_b_frag<FIN, FOUT, KK, NT * 16>(rootW, koff, col);
      acc[NT] = __builtin_amdgcn_wmma_f32_16x16x32_bf16(false, a, false, b,
                                                        (short)0, acc[NT], false, false);
    });
  });

  const int mhi = (lane >> 4) << 3;
  static_for<NTILES>([&](auto NT_) {
    constexpr int NT    = decltype(NT_)::value;
    constexpr bool full = (NT * 16 + 15 < FOUT);
    int cc = NT * 16 + col;
    if (full || cc < FOUT) {
      float bv = bias[cc];
#pragma unroll
      for (int r = 0; r < 8; ++r) {
        int m = nb + r + mhi;
        if (m < N_NODES) out[(long)m * FOUT + cc] = acc[NT][r] + bv;
      }
    }
  });
}

// ---------- 4b) edge GEMM: 16-pair tiles within one kernel bin, fused gather,
//              scale folded into epilogue, atomic scatter-add to dst -----------
template<int KTILES, int NTILES, int FIN, int FOUT, int KSTRIDE>
__global__ void edge_gemm_t(const __bf16* __restrict__ hb,
                            const float* __restrict__ W,
                            const int* __restrict__ ei,          // [2, E]
                            const int* __restrict__ pair_edge,
                            const float* __restrict__ pair_bas,
                            const int* __restrict__ tileK,
                            const int* __restrict__ tileP,
                            const int* __restrict__ n_tiles,
                            float* __restrict__ out) {
  __shared__ int   sDst[16];
  __shared__ float sScale[16];
  if ((int)blockIdx.x >= n_tiles[0]) return;   // uniform per block
  const int lane   = threadIdx.x;
  const int k      = tileK[blockIdx.x];
  const int tp     = tileP[blockIdx.x];
  const int pstart = tp & 0xFFFFFF;
  const int cnt    = ((unsigned)tp) >> 24;
  const int row    = lane & 15;

  // per-row metadata (both half-waves read the same 16 rows; L1 hits)
  int   e    = (row < cnt) ? pair_edge[pstart + row] : 0;
  int   srow = (row < cnt) ? ei[N_EDGES + e] : 0;     // src node, clamp invalid->0
  if (lane < 16) {
    sDst[lane]   = (lane < cnt) ? ei[e] : -1;         // dst node or invalid
    sScale[lane] = (lane < cnt) ? pair_bas[pstart + lane] : 0.0f;
  }
  __syncthreads();

  const float* Wk = W + (long)k * (FIN * FOUT);
  const __bf16* rowp = hb + (long)srow * KSTRIDE;
  const int abase = (lane < 16) ? 0 : 8;
  const int koff  = (lane < 16) ? 0 : 8;
  const int col   = lane & 15;

  v8f acc[NTILES];
#pragma unroll
  for (int nt = 0; nt < NTILES; ++nt) acc[nt] = {};

  static_for<KTILES>([&](auto KT_) {
    constexpr int KK = decltype(KT_)::value * 32;
    v16bf a = load_a_frag(rowp, KK, abase);  // raw features; basis scale in epilogue
    static_for<NTILES>([&](auto NT_) {
      constexpr int NT = decltype(NT_)::value;
      v16bf b = load_b_frag<FIN, FOUT, KK, NT * 16>(Wk, koff, col);
      acc[NT] = __builtin_amdgcn_wmma_f32_16x16x32_bf16(false, a, false, b,
                                                        (short)0, acc[NT], false, false);
    });
  });

  const int mhi = (lane >> 4) << 3;
#pragma unroll
  for (int r = 0; r < 8; ++r) {
    const int   M = r + mhi;
    const int   d = sDst[M];
    const float s = sScale[M];
    if (d >= 0) {
      static_for<NTILES>([&](auto NT_) {
        constexpr int NT    = decltype(NT_)::value;
        constexpr bool full = (NT * 16 + 15 < FOUT);
        int cc = NT * 16 + col;
        if (full || cc < FOUT) atomicAdd(&out[(long)d * FOUT + cc], acc[NT][r] * s);
      });
    }
  }
}

// ---------- 5) fused ELU + eval-BN, writes next layer's padded bf16 activations ----------
__global__ void elu_bn_kernel(const float* __restrict__ conv, int fout, int kstride,
                              const float* __restrict__ g, const float* __restrict__ be,
                              const float* __restrict__ rm, const float* __restrict__ rv,
                              __bf16* __restrict__ out) {
  int idx = blockIdx.x * blockDim.x + threadIdx.x;
  if (idx >= N_NODES * kstride) return;
  int n = idx / kstride, k = idx - n * kstride;
  float h = 0.0f;
  if (k < fout) {
    float v = conv[n * fout + k];
    float e = v > 0.0f ? v : expm1f(v);
    h = (e - rm[k]) * (g[k] * rsqrtf(rv[k] + BN_EPS)) + be[k];
  }
  out[idx] = f2bf(h);
}

// =====================================================================
extern "C" void kernel_launch(void* const* d_in, const int* in_sizes, int n_in,
                              void* d_out, int out_size, void* d_ws, size_t ws_size,
                              hipStream_t stream) {
  const float* x    = (const float*)d_in[0];
  const int*   ei   = (const int*)d_in[1];
  const float* attr = (const float*)d_in[2];
  const float* w[4]    = {(const float*)d_in[3], (const float*)d_in[6],
                          (const float*)d_in[9], (const float*)d_in[12]};
  const float* root[4] = {(const float*)d_in[4], (const float*)d_in[7],
                          (const float*)d_in[10], (const float*)d_in[13]};
  const float* bias[4] = {(const float*)d_in[5], (const float*)d_in[8],
                          (const float*)d_in[11], (const float*)d_in[14]};
  const float* g[3]  = {(const float*)d_in[15], (const float*)d_in[19], (const float*)d_in[23]};
  const float* be[3] = {(const float*)d_in[16], (const float*)d_in[20], (const float*)d_in[24]};
  const float* rm[3] = {(const float*)d_in[17], (const float*)d_in[21], (const float*)d_in[25]};
  const float* rv[3] = {(const float*)d_in[18], (const float*)d_in[22], (const float*)d_in[26]};

  // ---- carve workspace ----
  char* p = (char*)d_ws;
  auto carve = [&](size_t bytes) -> void* {
    void* r = (void*)p;
    p += (bytes + 255) & ~(size_t)255;
    return r;
  };
  int*    wiBuf      = (int*)   carve((size_t)PAIRS * 4);
  float*  basBuf     = (float*) carve((size_t)PAIRS * 4);
  int*    pair_edge  = (int*)   carve((size_t)PAIRS * 4);
  float*  pair_bas   = (float*) carve((size_t)PAIRS * 4);
  int*    bin_cnt    = (int*)   carve((size_t)KTOT * 4);
  int*    bin_start  = (int*)   carve((size_t)KTOT * 4);
  int*    bin_cur    = (int*)   carve((size_t)KTOT * 4);
  int*    tile_start = (int*)   carve((size_t)KTOT * 4);
  int*    tileK      = (int*)   carve((size_t)MAX_TILES * 4);
  int*    tileP      = (int*)   carve((size_t)MAX_TILES * 4);
  int*    n_tiles    = (int*)   carve(256);
  __bf16* hA         = (__bf16*)carve((size_t)N_NODES * 128 * 2);
  __bf16* hB         = (__bf16*)carve((size_t)N_NODES * 128 * 2);
  float*  convO      = (float*) carve((size_t)N_NODES * 100 * 4);

  hipError_t rc;
  rc = hipMemsetAsync(bin_cnt, 0, (size_t)KTOT * 4, stream); (void)rc;
  rc = hipMemsetAsync(bin_cur, 0, (size_t)KTOT * 4, stream); (void)rc;

  // ---- basis + counting sort by kernel index ----
  basis_kernel<<<(N_EDGES + 255) / 256, 256, 0, stream>>>(attr, wiBuf, basBuf);
  hist_kernel<<<(PAIRS + 255) / 256, 256, 0, stream>>>(wiBuf, bin_cnt);
  scan_kernel<<<1, 1024, 0, stream>>>(bin_cnt, bin_start, tile_start, n_tiles);
  tile_fill_kernel<<<(KTOT + 255) / 256, 256, 0, stream>>>(bin_cnt, bin_start,
                                                           tile_start, tileK, tileP);
  scatter_sort_kernel<<<(PAIRS + 255) / 256, 256, 0, stream>>>(wiBuf, basBuf, bin_start,
                                                               bin_cur, pair_edge, pair_bas);

  convert_x_kernel<<<(N_NODES * 64 + 255) / 256, 256, 0, stream>>>(x, 50, 64, hA);

  // ---- Layer 1: 50 -> 75 (Kpad 64, KT 2, NT 5) ----
  root_gemm_t<2, 5, 50, 75, 64><<<MTILES, 32, 0, stream>>>(hA, root[0], bias[0], convO);
  edge_gemm_t<2, 5, 50, 75, 64><<<MAX_TILES, 32, 0, stream>>>(
      hA, w[0], ei, pair_edge, pair_bas, tileK, tileP, n_tiles, convO);
  elu_bn_kernel<<<(N_NODES * 96 + 255) / 256, 256, 0, stream>>>(
      convO, 75, 96, g[0], be[0], rm[0], rv[0], hB);

  // ---- Layer 2: 75 -> 100 (Kpad 96, KT 3, NT 7) ----
  root_gemm_t<3, 7, 75, 100, 96><<<MTILES, 32, 0, stream>>>(hB, root[1], bias[1], convO);
  edge_gemm_t<3, 7, 75, 100, 96><<<MAX_TILES, 32, 0, stream>>>(
      hB, w[1], ei, pair_edge, pair_bas, tileK, tileP, n_tiles, convO);
  elu_bn_kernel<<<(N_NODES * 128 + 255) / 256, 256, 0, stream>>>(
      convO, 100, 128, g[1], be[1], rm[1], rv[1], hA);

  // ---- Layer 3: 100 -> 75 (Kpad 128, KT 4, NT 5) ----
  root_gemm_t<4, 5, 100, 75, 128><<<MTILES, 32, 0, stream>>>(hA, root[2], bias[2], convO);
  edge_gemm_t<4, 5, 100, 75, 128><<<MAX_TILES, 32, 0, stream>>>(
      hA, w[2], ei, pair_edge, pair_bas, tileK, tileP, n_tiles, convO);
  elu_bn_kernel<<<(N_NODES * 96 + 255) / 256, 256, 0, stream>>>(
      convO, 75, 96, g[2], be[2], rm[2], rv[2], hB);

  // ---- Layer 4: 75 -> 50 (Kpad 96, KT 3, NT 4), conv out = d_out ----
  root_gemm_t<3, 4, 75, 50, 96><<<MTILES, 32, 0, stream>>>(hB, root[3], bias[3], (float*)d_out);
  edge_gemm_t<3, 4, 75, 50, 96><<<MAX_TILES, 32, 0, stream>>>(
      hB, w[3], ei, pair_edge, pair_bas, tileK, tileP, n_tiles, (float*)d_out);

  (void)in_sizes; (void)n_in; (void)out_size; (void)ws_size;
}
